// GATLayer_57543971832009
// MI455X (gfx1250) — matrix operand order repaired and checked by reading, not compiled, not run on hardware
//
#include <hip/hip_runtime.h>

// ---------------------------------------------------------------------------
// Masked multi-head attention (GAT-style) for MI455X / gfx1250.
// wave32 WMMA (bf16, f32 acc) + Tensor Data Mover staging + flash softmax.
// B=4, N=2048, IN_F=OUT_F=256, HEADS=4, HEAD_DIM=64.
// ---------------------------------------------------------------------------

#define BB   4
#define NN   2048
#define CF   256
#define HDS  4
#define DH   64
#define MROWS (BB * NN)          // 8192 rows of h / output

typedef __bf16 bf16_t;
typedef bf16_t v16bf __attribute__((ext_vector_type(16)));
typedef bf16_t v8bf  __attribute__((ext_vector_type(8)));
typedef float  v8f   __attribute__((ext_vector_type(8)));
typedef unsigned short us8 __attribute__((ext_vector_type(8)));
typedef unsigned int v4u __attribute__((ext_vector_type(4)));
typedef int v8i __attribute__((ext_vector_type(8)));
typedef int v4i __attribute__((ext_vector_type(4)));

// float -> bf16 (round to nearest even), stored as ushort
__device__ __forceinline__ unsigned short f2bf_u(float x) {
  unsigned u = __builtin_bit_cast(unsigned, x);
  unsigned r = u + 0x7FFFu + ((u >> 16) & 1u);
  return (unsigned short)(r >> 16);
}

// 16-byte load of 8 bf16 values (global or LDS via generic pointer)
__device__ __forceinline__ v8bf ld8(const unsigned short* p) {
  us8 t = *(const us8*)p;
  return __builtin_bit_cast(v8bf, t);
}

__device__ __forceinline__ v16bf cat(v8bf a, v8bf b) {
  return __builtin_shufflevector(a, b, 0,1,2,3,4,5,6,7,8,9,10,11,12,13,14,15);
}

__device__ __forceinline__ v8f bwmma(v16bf a, v16bf b, v8f c) {
  return __builtin_amdgcn_wmma_f32_16x16x32_bf16(false, a, false, b,
                                                 (short)0, c, false, false);
}

// The TDM intrinsic is not modeled as writing LDS, so the optimizer would
// prove the shared array store-free and fold its loads to undef. Escape the
// LDS pointer into a volatile asm with a memory clobber (a potential writer),
// but DISCARD the laundered pointer: subsequent loads go through the original
// smem pointer, keeping addrspace(3) provenance -> ds_load_b128, not flat.
__device__ __forceinline__ void lds_mark_written(unsigned short* p) {
  asm volatile("" : "+v"(p) :: "memory");
}

// ---------------------------------------------------------------------------
// Tensor Data Mover: 2D bf16 tile (rows of 256 elements, contiguous, row
// stride 256) from global -> LDS, with hardware LDS padding of 16B per 512B
// row so LDS row stride becomes 264 ushorts (bank-conflict-free b128 reads).
// D# encoding per CDNA5 ISA 8.3/8.4.  6-arg builtin on this toolchain:
//   (uint32x4 g0, int32x8 g1, int32x4 g2, int32x4 g3, int32x8 g4, i32 cpol)
// ---------------------------------------------------------------------------
__device__ __forceinline__ void tdm_load_rows256(unsigned lds_byte_off,
                                                 const void* gaddr,
                                                 unsigned tensor_rows,
                                                 unsigned tile_rows) {
  unsigned long long ga = (unsigned long long)gaddr;
  v4u g0;
  g0.x = 0x1u;                                      // count=1 (valid), user mode
  g0.y = lds_byte_off;                              // lds_addr
  g0.z = (unsigned)(ga & 0xFFFFFFFFu);              // global_addr[31:0]
  g0.w = (unsigned)((ga >> 32) & 0x01FFFFFFu) | (2u << 30);  // [56:32] | type=2
  v8i g1;
  g1[0] = (int)((1u << 16)        // data_size = 2 bytes
              | (1u << 20)        // pad_enable
              | (6u << 22)        // pad_interval: 128 dwords (512B rows)
              | (3u << 25));      // pad_amount: 4 dwords (16B)
  g1[1] = (int)((256u & 0xFFFFu) << 16);                       // tensor_dim0 lo
  g1[2] = (int)((tensor_rows & 0xFFFFu) << 16);                // dim0 hi=0 | dim1 lo
  g1[3] = (int)(((tensor_rows >> 16) & 0xFFFFu) | (256u << 16)); // dim1 hi | tile_dim0
  g1[4] = (int)(tile_rows & 0xFFFFu);                          // tile_dim1, tile_dim2=0
  g1[5] = 256;                                                 // tensor_dim0_stride lo
  g1[6] = 0;
  g1[7] = 0;
  v4i g2 = {0, 0, 0, 0};
  v4i g3 = {0, 0, 0, 0};
  v8i g4 = {0, 0, 0, 0, 0, 0, 0, 0};
  __builtin_amdgcn_tensor_load_to_lds(g0, g1, g2, g3, g4, 0);
}

// ---------------------------------------------------------------------------
// Prep: fp32 -> bf16 conversions done once (so GEMM staging is pure DMA).
// ---------------------------------------------------------------------------
__global__ __launch_bounds__(256) void cvt_h_kernel(
    const float* __restrict__ h, unsigned short* __restrict__ hb, int n) {
  for (int i = blockIdx.x * 256 + threadIdx.x; i < n; i += gridDim.x * 256)
    hb[i] = f2bf_u(h[i]);
}

// WT[n][k] = bf16(W[k][n])  (pre-transposed weights for B-operand tiles)
__global__ __launch_bounds__(256) void cvt_wT_kernel(
    const float* __restrict__ W, unsigned short* __restrict__ WT) {
  int k = blockIdx.x;
  int n = threadIdx.x;
  WT[n * CF + k] = f2bf_u(W[k * CF + n]);
}

// ---------------------------------------------------------------------------
// Kernel 1: QKV projections.  q,k -> [bh][n][64] bf16 ; v -> [bh][d][n] bf16.
// Block = 256 threads (8 waves): 16(M) x 128(N) tile of one matrix.
// A/B tiles staged to LDS by the TDM; waves do ds_load_b128 + WMMA only.
// ---------------------------------------------------------------------------
#define AS 264   // LDS row stride in ushorts (512B row + 16B TDM pad)

__global__ __launch_bounds__(256) void qkv_proj(
    const unsigned short* __restrict__ hb,
    const unsigned short* __restrict__ WqT, const unsigned short* __restrict__ WkT,
    const unsigned short* __restrict__ WvT,
    unsigned short* __restrict__ qb, unsigned short* __restrict__ kb,
    unsigned short* __restrict__ vT) {
  __shared__ __align__(16) unsigned short smem[16 * AS + 128 * AS];
  const int t = threadIdx.x;
  const int row0 = blockIdx.x * 16;
  const int n0 = blockIdx.y * 128;
  const int mat = blockIdx.z;
  const unsigned short* __restrict__ WT = (mat == 0) ? WqT : (mat == 1) ? WkT : WvT;

  // wave 0 issues both TDM descriptors (scalar-uniform branch: TDM ignores EXEC)
  if (__builtin_amdgcn_readfirstlane(t) == 0) {
    tdm_load_rows256(0u, hb + (size_t)row0 * CF, MROWS, 16);
    tdm_load_rows256(16u * AS * 2u, WT + (size_t)n0 * CF, CF, 128);
    __builtin_amdgcn_s_wait_tensorcnt(0);
  }
  __syncthreads();
  lds_mark_written(smem);   // TDM wrote LDS; keep ds_loads alive + AS3 path

  const unsigned short* sA = smem;            // h tile  [16 m][256 k], stride AS
  const unsigned short* sB = smem + 16 * AS;  // WT tile [128 n][256 k], stride AS

  const int wave = t >> 5, l = t & 31;
  const int m = l & 15, half = l >> 4;
  v8f c = {};
  const unsigned short* aRow = &sA[m * AS];
  const unsigned short* bRow = &sB[(wave * 16 + m) * AS];
  #pragma unroll
  for (int s = 0; s < 8; ++s) {
    int k0 = s * 32;
    v16bf a = cat(ld8(aRow + k0 + 8 * half), ld8(aRow + k0 + 16 + 8 * half));
    v16bf b = cat(ld8(bRow + k0 + 8 * half), ld8(bRow + k0 + 16 + 8 * half));
    c = bwmma(a, b, c);
  }
  #pragma unroll
  for (int r = 0; r < 8; ++r) {
    int row = row0 + r + 8 * half;            // b*N + n
    int col = n0 + wave * 16 + m;             // head*64 + d
    int bi = row >> 11, n = row & (NN - 1);
    int head = col >> 6, d = col & (DH - 1);
    unsigned short val = f2bf_u(c[r]);
    int bh = bi * HDS + head;
    if (mat == 0)      qb[((size_t)bh * NN + n) * DH + d] = val;
    else if (mat == 1) kb[((size_t)bh * NN + n) * DH + d] = val;
    else               vT[((size_t)bh * DH + d) * NN + n] = val;
  }
}

// ---------------------------------------------------------------------------
// Kernel 2: fused masked attention (flash style, online softmax).
// One wave per (bh, 16-row Q tile); 64 blocks of 32 key columns.
// ---------------------------------------------------------------------------
__global__ __launch_bounds__(256) void attn_kernel(
    const unsigned short* __restrict__ qb, const unsigned short* __restrict__ kb,
    const unsigned short* __restrict__ vT, const int* __restrict__ adj,
    unsigned short* __restrict__ attnb) {
  __shared__ __align__(16) unsigned short plds[8][16 * 32];  // per-wave P scratch
  const int t = threadIdx.x;
  const int wave = t >> 5, l = t & 31;
  const int jid = blockIdx.x * 8 + wave;   // 0..2047
  const int bh = jid >> 7;                 // 0..15
  const int qn0 = (jid & 127) * 16;        // query row base
  const int m = l & 15, half = l >> 4;
  const int bi = bh >> 2, hh = bh & 3;

  const unsigned short* qrow = qb + ((size_t)bh * NN + qn0 + m) * DH;
  const v16bf qa0 = cat(ld8(qrow + 16 * half),      ld8(qrow + 16 * half + 8));
  const v16bf qa1 = cat(ld8(qrow + 32 + 16 * half), ld8(qrow + 32 + 16 * half + 8));

  v8f O0 = {}, O1 = {}, O2 = {}, O3 = {};
  float mrun[8], lrun[8];
  #pragma unroll
  for (int r = 0; r < 8; ++r) { mrun[r] = -3.0e38f; lrun[r] = 0.0f; }
  unsigned short* pw = &plds[wave][0];

  for (int ct = 0; ct < NN / 32; ++ct) {
    const int cb = ct * 32;
    // prefetch adjacency rows 8 blocks ahead (lowers to global_prefetch_b8)
    if (ct + 8 < NN / 32)
      __builtin_prefetch((const void*)(adj + (qn0 + m) * NN + cb + 8 * 32), 0, 1);

    v8f S0 = {}, S1 = {};
    {
      const unsigned short* kr0 = kb + ((size_t)bh * NN + cb + m) * DH;
      const unsigned short* kr1 = kr0 + 16 * DH;
      v16bf b00 = cat(ld8(kr0 + 16 * half),      ld8(kr0 + 16 * half + 8));
      v16bf b01 = cat(ld8(kr0 + 32 + 16 * half), ld8(kr0 + 32 + 16 * half + 8));
      v16bf b10 = cat(ld8(kr1 + 16 * half),      ld8(kr1 + 16 * half + 8));
      v16bf b11 = cat(ld8(kr1 + 32 + 16 * half), ld8(kr1 + 32 + 16 * half + 8));
      S0 = bwmma(qa0, b00, S0);  S0 = bwmma(qa1, b01, S0);
      S1 = bwmma(qa0, b10, S1);  S1 = bwmma(qa1, b11, S1);
    }
    // scale (1/sqrt(64)) + adjacency mask
    #pragma unroll
    for (int r = 0; r < 8; ++r) {
      int row = qn0 + r + 8 * half;
      int a0 = adj[row * NN + cb + m];
      int a1 = adj[row * NN + cb + 16 + m];
      S0[r] = (a0 == 0) ? -1.0e9f : S0[r] * 0.125f;
      S1[r] = (a1 == 0) ? -1.0e9f : S1[r] * 0.125f;
    }
    // online softmax update + write P (bf16) to per-wave LDS in C-layout
    #pragma unroll
    for (int r = 0; r < 8; ++r) {
      float vmax = fmaxf(S0[r], S1[r]);
      #pragma unroll
      for (int off = 1; off < 16; off <<= 1)
        vmax = fmaxf(vmax, __shfl_xor(vmax, off, 16));
      float mnew  = fmaxf(mrun[r], vmax);
      float alpha = __expf(mrun[r] - mnew);
      float p0 = __expf(S0[r] - mnew);
      float p1 = __expf(S1[r] - mnew);
      float rs = p0 + p1;
      #pragma unroll
      for (int off = 1; off < 16; off <<= 1)
        rs += __shfl_xor(rs, off, 16);
      lrun[r] = lrun[r] * alpha + rs;
      mrun[r] = mnew;
      O0[r] *= alpha; O1[r] *= alpha; O2[r] *= alpha; O3[r] *= alpha;
      int prow = r + 8 * half;
      pw[prow * 32 + m]      = f2bf_u(p0);
      pw[prow * 32 + 16 + m] = f2bf_u(p1);
    }
    // same-wave DS ordering (C-layout stores -> A-layout loads)
    asm volatile("s_wait_dscnt 0" ::: "memory");
    v16bf pa = cat(ld8(pw + m * 32 + 16 * half), ld8(pw + m * 32 + 16 * half + 8));
    // O += P @ V  (V transposed: contiguous 16-key runs per d row)
    const unsigned short* vb = vT + ((size_t)bh * DH + m) * NN + cb + 16 * half;
    v16bf v0 = cat(ld8(vb + 0 * 16 * NN), ld8(vb + 0 * 16 * NN + 8));
    v16bf v1 = cat(ld8(vb + 1 * 16 * NN), ld8(vb + 1 * 16 * NN + 8));
    v16bf v2 = cat(ld8(vb + 2 * 16 * NN), ld8(vb + 2 * 16 * NN + 8));
    v16bf v3 = cat(ld8(vb + 3 * 16 * NN), ld8(vb + 3 * 16 * NN + 8));
    O0 = bwmma(pa, v0, O0);
    O1 = bwmma(pa, v1, O1);
    O2 = bwmma(pa, v2, O2);
    O3 = bwmma(pa, v3, O3);
  }

  #pragma unroll
  for (int r = 0; r < 8; ++r) {
    float inv = 1.0f / lrun[r];
    int row = qn0 + r + 8 * half;
    unsigned short* orow = attnb + ((size_t)bi * NN + row) * CF + hh * DH + m;
    orow[0]  = f2bf_u(O0[r] * inv);
    orow[16] = f2bf_u(O1[r] * inv);
    orow[32] = f2bf_u(O2[r] * inv);
    orow[48] = f2bf_u(O3[r] * inv);
  }
}

// ---------------------------------------------------------------------------
// Kernel 3: out = h + attn @ Wo + bo   (fp32 output); TDM-staged tiles.
// ---------------------------------------------------------------------------
__global__ __launch_bounds__(256) void out_proj(
    const unsigned short* __restrict__ attnb, const unsigned short* __restrict__ WoT,
    const float* __restrict__ bo, const float* __restrict__ h,
    float* __restrict__ out) {
  __shared__ __align__(16) unsigned short smem[16 * AS + 128 * AS];
  const int t = threadIdx.x;
  const int row0 = blockIdx.x * 16;
  const int n0 = blockIdx.y * 128;

  if (__builtin_amdgcn_readfirstlane(t) == 0) {
    tdm_load_rows256(0u, attnb + (size_t)row0 * CF, MROWS, 16);
    tdm_load_rows256(16u * AS * 2u, WoT + (size_t)n0 * CF, CF, 128);
    __builtin_amdgcn_s_wait_tensorcnt(0);
  }
  __syncthreads();
  lds_mark_written(smem);

  const unsigned short* sA = smem;
  const unsigned short* sB = smem + 16 * AS;

  const int wave = t >> 5, l = t & 31;
  const int m = l & 15, half = l >> 4;
  v8f c = {};
  const unsigned short* aRow = &sA[m * AS];
  const unsigned short* bRow = &sB[(wave * 16 + m) * AS];
  #pragma unroll
  for (int s = 0; s < 8; ++s) {
    int k0 = s * 32;
    v16bf a = cat(ld8(aRow + k0 + 8 * half), ld8(aRow + k0 + 16 + 8 * half));
    v16bf b = cat(ld8(bRow + k0 + 8 * half), ld8(bRow + k0 + 16 + 8 * half));
    c = bwmma(a, b, c);
  }
  #pragma unroll
  for (int r = 0; r < 8; ++r) {
    int row = row0 + r + 8 * half;
    int col = n0 + wave * 16 + m;
    out[row * CF + col] = h[row * CF + col] + c[r] + bo[col];
  }
}

// ---------------------------------------------------------------------------
extern "C" void kernel_launch(void* const* d_in, const int* in_sizes, int n_in,
                              void* d_out, int out_size, void* d_ws,
                              size_t ws_size, hipStream_t stream) {
  const float* h   = (const float*)d_in[0];
  const int*   adj = (const int*)d_in[1];
  const float* Wq  = (const float*)d_in[2];
  const float* Wk  = (const float*)d_in[3];
  const float* Wv  = (const float*)d_in[4];
  const float* Wo  = (const float*)d_in[5];
  const float* bo  = (const float*)d_in[6];
  float* out = (float*)d_out;

  const size_t ELEMS = (size_t)BB * HDS * NN * DH;  // 2M bf16 per tensor
  unsigned short* qb    = (unsigned short*)d_ws;
  unsigned short* kb    = qb + ELEMS;
  unsigned short* vT    = kb + ELEMS;
  unsigned short* attnb = vT + ELEMS;               // [B*N][256] bf16
  unsigned short* hb    = attnb + ELEMS;            // [B*N][256] bf16
  unsigned short* WqT   = hb + ELEMS;
  unsigned short* WkT   = WqT + (size_t)CF * CF;
  unsigned short* WvT   = WkT + (size_t)CF * CF;
  unsigned short* WoT   = WvT + (size_t)CF * CF;

  cvt_h_kernel<<<dim3(2048), 256, 0, stream>>>(h, hb, (int)(MROWS * CF));
  cvt_wT_kernel<<<dim3(256), 256, 0, stream>>>(Wq, WqT);
  cvt_wT_kernel<<<dim3(256), 256, 0, stream>>>(Wk, WkT);
  cvt_wT_kernel<<<dim3(256), 256, 0, stream>>>(Wv, WvT);
  cvt_wT_kernel<<<dim3(256), 256, 0, stream>>>(Wo, WoT);

  qkv_proj<<<dim3(512, 2, 3), 256, 0, stream>>>(hb, WqT, WkT, WvT, qb, kb, vT);
  attn_kernel<<<dim3(256), 256, 0, stream>>>(qb, kb, vT, adj, attnb);
  out_proj<<<dim3(512, 2), 256, 0, stream>>>(attnb, WoT, bo, h, out);
}
